// ST_layer_22179211116814
// MI455X (gfx1250) — matrix operand order repaired and checked by reading, not compiled, not run on hardware
//
#include <hip/hip_runtime.h>
#include <math.h>
#include <stdint.h>

// ---------------------------------------------------------------------------
// Problem constants (match reference)
// ---------------------------------------------------------------------------
#define BB   8
#define TT   64
#define NN   1024
#define DD   2
#define HH   128
#define DMM  128      // T*D
#define CC   2048     // N*D
#define EPSF 1e-6f

typedef __attribute__((ext_vector_type(16))) __bf16 v16bf;
typedef __attribute__((ext_vector_type(8)))  float  v8f;

__device__ __forceinline__ __bf16 tob(float f) { return (__bf16)f; }

__device__ __forceinline__ v8f vzero8() {
    v8f z;
#pragma unroll
    for (int i = 0; i < 8; ++i) z[i] = 0.f;
    return z;
}

// A fragment: 16x32 bf16, rows m0..m0+15 of LDS tile `base` (row stride ld).
// Layout (ISA 7.12.2, 16-bit A 16x32): lane = 16*h + m;
//   VGPR v in 0..3: K = 2v + 8h + {0,1};  v in 4..7: K = 16 + 2(v-4) + 8h + {0,1}
__device__ __forceinline__ v16bf load_frag_a(const __bf16* base, int ld) {
    const int lane = threadIdx.x & 31;
    const int m = lane & 15, h = lane >> 4;
    const __bf16* row = base + m * ld;
    v16bf a;
#pragma unroll
    for (int v = 0; v < 8; ++v) {
        int k = ((v < 4) ? (2 * v) : (16 + 2 * (v - 4))) + 8 * h;
        a[2 * v]     = row[k];
        a[2 * v + 1] = row[k + 1];
    }
    return a;
}

// B fragment: 32x16 bf16 from LDS tile (row stride ld), column offset n0.
__device__ __forceinline__ v16bf load_frag_b(const __bf16* base, int ld, int n0) {
    const int lane = threadIdx.x & 31;
    const int h = lane >> 4;
    const __bf16* col = base + n0 + (lane & 15);
    v16bf b;
#pragma unroll
    for (int j = 0; j < 16; ++j) b[j] = col[(((j < 8) ? j : (8 + j)) + 8 * h) * ld];
    return b;
}

__device__ __forceinline__ v8f wmma_bf16(v16bf a, v16bf b, v8f c) {
    return __builtin_amdgcn_wmma_f32_16x16x32_bf16(
        /*neg_a=*/false, a, /*neg_b=*/false, b,
        /*c_mod=*/(short)0, c, /*reuse_a=*/false, /*reuse_b=*/false);
}

// Async 16B global -> LDS copy (gfx1250, ASYNCcnt-tracked).
// VDST VGPR carries the wave-relative LDS byte offset (= low 32 bits of the
// generic LDS pointer on CDNA).
__device__ __forceinline__ void async_copy16(const void* lds_ptr, const void* gptr) {
    uint32_t lds_off = (uint32_t)(uintptr_t)lds_ptr;
    asm volatile("global_load_async_to_lds_b128 %0, %1, off"
                 :: "v"(lds_off), "v"(gptr) : "memory");
}
__device__ __forceinline__ void wait_async0() {
    asm volatile("s_wait_asynccnt 0x0" ::: "memory");
}

// ---------------------------------------------------------------------------
// 1) GCN: per (b,d,n) row: relu-MLP over time axis, +x residual -> gbuf[B,T,N,D]
// ---------------------------------------------------------------------------
__global__ __launch_bounds__(256)
void gcn_kernel(const float* __restrict__ x, const float* __restrict__ graph,
                const float* __restrict__ W1, const float* __restrict__ b1,
                const float* __restrict__ W2, const float* __restrict__ b2,
                float* __restrict__ gbuf) {
    __shared__ float sW1[TT * HH];
    __shared__ float sW2[HH * TT];
    __shared__ float sb1[HH];
    __shared__ float sb2[TT];
    for (int i = threadIdx.x; i < TT * HH; i += 256) sW1[i] = W1[i];
    for (int i = threadIdx.x; i < HH * TT; i += 256) sW2[i] = W2[i];
    if (threadIdx.x < HH) sb1[threadIdx.x] = b1[threadIdx.x];
    if (threadIdx.x < TT) sb2[threadIdx.x] = b2[threadIdx.x];
    __syncthreads();

    int row = blockIdx.x * 256 + threadIdx.x;     // over B*D*N = 16384
    int b   = row >> 11;
    int rem = row & 2047;
    int d   = rem >> 10;
    int n   = rem & 1023;
    float dgn = graph[(size_t)n * NN + n];

    float xr[TT];
#pragma unroll
    for (int t = 0; t < TT; ++t)
        xr[t] = x[(((size_t)b * TT + t) * NN + n) * DD + d];

    float o[TT];
#pragma unroll
    for (int t = 0; t < TT; ++t) o[t] = sb2[t];

    for (int j = 0; j < HH; ++j) {
        float u = sb1[j];
#pragma unroll
        for (int t = 0; t < TT; ++t) u += xr[t] * sW1[t * HH + j];
        u *= dgn;
        float f = u > 0.f ? u : 0.f;
#pragma unroll
        for (int t = 0; t < TT; ++t) o[t] += f * sW2[j * TT + t];
    }
#pragma unroll
    for (int t = 0; t < TT; ++t) {
        float g = o[t] * dgn;
        g = g > 0.f ? g : 0.f;
        gbuf[(((size_t)b * TT + t) * NN + n) * DD + d] = g + xr[t];
    }
}

// ---------------------------------------------------------------------------
// 2) QKV projections: xf[8192,128] @ {Wq,Wk,Wv} (+bias), relu^2 feature map,
//    sin/cos split -> q_[.,256], k_[.,256], v_[.,128].  WMMA bf16.
// ---------------------------------------------------------------------------
__global__ __launch_bounds__(256)
void qkv_kernel(const float* __restrict__ x,
                const float* __restrict__ Wq, const float* __restrict__ bq,
                const float* __restrict__ Wk, const float* __restrict__ bk,
                const float* __restrict__ Wv, const float* __restrict__ bv,
                float* __restrict__ q_, float* __restrict__ k_,
                float* __restrict__ v_) {
    __shared__ __bf16 sA[64][136];
    __shared__ __bf16 sW[128][136];

    const int row0 = blockIdx.x * 64;         // over 8192 rows (b*1024+n)
    const int wave = threadIdx.x >> 5;
    const int lane = threadIdx.x & 31;

    for (int i = threadIdx.x; i < 64 * 128; i += 256) {
        int r = i >> 7, c = i & 127;
        int row = row0 + r;
        int bb = row >> 10, n = row & 1023;
        int t = c >> 1, d = c & 1;
        sA[r][c] = tob(x[(((size_t)bb * TT + t) * NN + n) * DD + d]);
    }

    const float* Wp[3] = {Wq, Wk, Wv};
    const float* bp[3] = {bq, bk, bv};
    const float PI_F = 3.14159265358979323846f;

    for (int p = 0; p < 3; ++p) {
        __syncthreads();
        for (int i = threadIdx.x; i < 128 * 128; i += 256)
            sW[i >> 7][i & 127] = tob(Wp[p][i]);
        __syncthreads();

        if (wave < 4) {
            v8f acc[8];
#pragma unroll
            for (int nt = 0; nt < 8; ++nt) acc[nt] = vzero8();
#pragma unroll
            for (int k0 = 0; k0 < 128; k0 += 32) {
                v16bf a = load_frag_a(&sA[wave * 16][k0], 136);
#pragma unroll
                for (int nt = 0; nt < 8; ++nt) {
                    v16bf bfr = load_frag_b(&sW[k0][0], 136, nt * 16);
                    acc[nt] = wmma_bf16(a, bfr, acc[nt]);
                }
            }
            const int m = lane & 15, h = lane >> 4;
#pragma unroll
            for (int nt = 0; nt < 8; ++nt) {
#pragma unroll
                for (int r = 0; r < 8; ++r) {
                    int row = row0 + wave * 16 + r + 8 * h;
                    int col = nt * 16 + m;
                    float u = acc[nt][r] + bp[p][col];
                    if (p == 2) {
                        v_[(size_t)row * 128 + col] = u;
                    } else {
                        float f = u > 0.f ? u * u : 0.f;
                        int n = row & 1023;
                        float wi = PI_F * 0.5f * (float)(n + 1) * (1.0f / (float)NN);
                        float sv = sinf(wi), cv = cosf(wi);
                        float* dst = (p == 0) ? q_ : k_;
                        dst[(size_t)row * 256 + col]       = f * sv;
                        dst[(size_t)row * 256 + 128 + col] = f * cv;
                    }
                }
            }
        }
    }
}

// ---------------------------------------------------------------------------
// 3) ksum[b,256] = sum_n k_[b,n,:]
// ---------------------------------------------------------------------------
__global__ __launch_bounds__(256)
void ksum_kernel(const float* __restrict__ k_, float* __restrict__ ksum) {
    int bb = blockIdx.x;
    int d  = threadIdx.x;
    const float* base = k_ + (size_t)bb * NN * 256 + d;
    float s = 0.f;
    for (int n = 0; n < NN; ++n) s += base[(size_t)n * 256];
    ksum[bb * 256 + d] = s;
}

// ---------------------------------------------------------------------------
// 4) z[b,n] = 1 / max(dot(q_[b,n], ksum[b]), EPS)
// ---------------------------------------------------------------------------
__global__ __launch_bounds__(256)
void z_kernel(const float* __restrict__ q_, const float* __restrict__ ksum,
              float* __restrict__ zbuf) {
    int row = blockIdx.x * 256 + threadIdx.x;   // 8192
    int bb = row >> 10;
    const float* qr = q_ + (size_t)row * 256;
    const float* ks = ksum + bb * 256;
    float s = 0.f;
    for (int i = 0; i < 256; ++i) s += qr[i] * ks[i];
    s = fmaxf(s, EPSF);
    zbuf[row] = 1.0f / s;
}

// ---------------------------------------------------------------------------
// 5) kv[b] = k_[b]^T @ v[b]  :  [256 x 1024] x [1024 x 128].  WMMA bf16.
// ---------------------------------------------------------------------------
__global__ __launch_bounds__(256)
void kv_kernel(const float* __restrict__ k_, const float* __restrict__ v_,
               float* __restrict__ kvb) {
    __shared__ __bf16 sA[64][40];
    __shared__ __bf16 sB[32][136];
    const int m0 = blockIdx.x * 64;
    const int bb = blockIdx.y;
    const int wave = threadIdx.x >> 5, lane = threadIdx.x & 31;
    const float* kb = k_ + (size_t)bb * NN * 256;
    const float* vb = v_ + (size_t)bb * NN * 128;

    v8f acc[8];
#pragma unroll
    for (int nt = 0; nt < 8; ++nt) acc[nt] = vzero8();

    for (int k0 = 0; k0 < NN; k0 += 32) {
        __syncthreads();
        for (int i = threadIdx.x; i < 64 * 32; i += 256) {
            int m = i >> 5, kk = i & 31;
            sA[m][kk] = tob(kb[(size_t)(k0 + kk) * 256 + (m0 + m)]);
        }
        for (int i = threadIdx.x; i < 32 * 128; i += 256) {
            int kk = i >> 7, c = i & 127;
            sB[kk][c] = tob(vb[(size_t)(k0 + kk) * 128 + c]);
        }
        __syncthreads();
        if (wave < 4) {
            v16bf a = load_frag_a(&sA[wave * 16][0], 40);
#pragma unroll
            for (int nt = 0; nt < 8; ++nt) {
                v16bf bfr = load_frag_b(&sB[0][0], 136, nt * 16);
                acc[nt] = wmma_bf16(a, bfr, acc[nt]);
            }
        }
    }
    if (wave < 4) {
        const int m = lane & 15, h = lane >> 4;
#pragma unroll
        for (int nt = 0; nt < 8; ++nt)
#pragma unroll
            for (int r = 0; r < 8; ++r) {
                int mm = m0 + wave * 16 + r + 8 * h;
                int col = nt * 16 + m;
                kvb[((size_t)bb * 256 + mm) * 128 + col] = acc[nt][r];
            }
    }
}

// ---------------------------------------------------------------------------
// 6) attnb = (q_ @ kv[b]) * z + xf  :  [8192 x 256] x [256 x 128].  WMMA bf16.
// ---------------------------------------------------------------------------
__global__ __launch_bounds__(256)
void attn1_kernel(const float* __restrict__ q_, const float* __restrict__ kvb,
                  const float* __restrict__ zbuf, const float* __restrict__ x,
                  float* __restrict__ attnb) {
    __shared__ __bf16 sA[64][40];
    __shared__ __bf16 sB[32][136];
    const int row0 = blockIdx.x * 64;
    const int bb = row0 >> 10;
    const int wave = threadIdx.x >> 5, lane = threadIdx.x & 31;

    v8f acc[8];
#pragma unroll
    for (int nt = 0; nt < 8; ++nt) acc[nt] = vzero8();

    for (int k0 = 0; k0 < 256; k0 += 32) {
        __syncthreads();
        for (int i = threadIdx.x; i < 64 * 32; i += 256) {
            int r = i >> 5, kk = i & 31;
            sA[r][kk] = tob(q_[(size_t)(row0 + r) * 256 + k0 + kk]);
        }
        for (int i = threadIdx.x; i < 32 * 128; i += 256) {
            int kk = i >> 7, c = i & 127;
            sB[kk][c] = tob(kvb[((size_t)bb * 256 + k0 + kk) * 128 + c]);
        }
        __syncthreads();
        if (wave < 4) {
            v16bf a = load_frag_a(&sA[wave * 16][0], 40);
#pragma unroll
            for (int nt = 0; nt < 8; ++nt) {
                v16bf bfr = load_frag_b(&sB[0][0], 136, nt * 16);
                acc[nt] = wmma_bf16(a, bfr, acc[nt]);
            }
        }
    }
    if (wave < 4) {
        const int m = lane & 15, h = lane >> 4;
#pragma unroll
        for (int nt = 0; nt < 8; ++nt)
#pragma unroll
            for (int r = 0; r < 8; ++r) {
                int row = row0 + wave * 16 + r + 8 * h;
                int col = nt * 16 + m;
                int n = row & 1023, t = col >> 1, d = col & 1;
                float val = acc[nt][r] * zbuf[row];
                val += x[(((size_t)bb * TT + t) * NN + n) * DD + d];   // + xf
                attnb[(size_t)row * 128 + col] = val;
            }
    }
}

// ---------------------------------------------------------------------------
// 7) attn2 = attnb @ Wo + bo; gcm = attn2 + gbuf + x; write into xc[B,C,T].
// ---------------------------------------------------------------------------
__global__ __launch_bounds__(256)
void attn2_kernel(const float* __restrict__ attnb, const float* __restrict__ Wo,
                  const float* __restrict__ bo, const float* __restrict__ gbuf,
                  const float* __restrict__ x, float* __restrict__ xc) {
    __shared__ __bf16 sA[64][136];
    __shared__ __bf16 sW[128][136];
    const int row0 = blockIdx.x * 64;
    const int wave = threadIdx.x >> 5, lane = threadIdx.x & 31;

    for (int i = threadIdx.x; i < 64 * 128; i += 256)
        sA[i >> 7][i & 127] = tob(attnb[(size_t)(row0 + (i >> 7)) * 128 + (i & 127)]);
    for (int i = threadIdx.x; i < 128 * 128; i += 256)
        sW[i >> 7][i & 127] = tob(Wo[i]);
    __syncthreads();

    if (wave < 4) {
        v8f acc[8];
#pragma unroll
        for (int nt = 0; nt < 8; ++nt) acc[nt] = vzero8();
#pragma unroll
        for (int k0 = 0; k0 < 128; k0 += 32) {
            v16bf a = load_frag_a(&sA[wave * 16][k0], 136);
#pragma unroll
            for (int nt = 0; nt < 8; ++nt) {
                v16bf bfr = load_frag_b(&sW[k0][0], 136, nt * 16);
                acc[nt] = wmma_bf16(a, bfr, acc[nt]);
            }
        }
        const int m = lane & 15, h = lane >> 4;
#pragma unroll
        for (int nt = 0; nt < 8; ++nt)
#pragma unroll
            for (int r = 0; r < 8; ++r) {
                int row = row0 + wave * 16 + r + 8 * h;
                int col = nt * 16 + m;
                int bb = row >> 10, n = row & 1023, t = col >> 1, d = col & 1;
                size_t xi = (((size_t)bb * TT + t) * NN + n) * DD + d;
                float g = acc[nt][r] + bo[col] + gbuf[xi] + x[xi];
                xc[((size_t)bb * CC + (n * 2 + d)) * TT + t] = g;
            }
    }
}

// ---------------------------------------------------------------------------
// 8a) Weight repack: Wm[o,ci,tap] (fp32) -> wbf[o][kg] (bf16), kg-major where
//     kg<2048 -> tap1 (current t), kg>=2048 -> tap0 (t-dil).
// ---------------------------------------------------------------------------
__global__ __launch_bounds__(256)
void wconv_kernel(const float* __restrict__ Wm, __bf16* __restrict__ wbf) {
    size_t j = (size_t)blockIdx.x * 256 + threadIdx.x;  // 2048*4096 = 8388608
    int o  = (int)(j >> 12);
    int kg = (int)(j & 4095);
    int ci = kg & (CC - 1);
    int tap = (kg < CC) ? 1 : 0;
    wbf[j] = tob(Wm[((size_t)o * CC + ci) * 2 + tap]);
}

// ---------------------------------------------------------------------------
// 8b) MSTCN block: dense causal conv as GEMM [2048 x 4096] x [4096 x 64] per
//     batch.  BM=128 (all 8 waves compute), double-buffered LDS; A-tiles come
//     in via global_load_async_to_lds_b128 (ASYNCcnt), B-tiles are VALU-built
//     (fp32->bf16 + causal shift).  Fused bias+ReLU+depthwise+ReLU epilogue.
// ---------------------------------------------------------------------------
#define MST_BM    128
#define MST_ITERS 128   // 4096 / 32

__global__ __launch_bounds__(256)
void mstcn_kernel(const float* __restrict__ Xin, const __bf16* __restrict__ wbf,
                  const float* __restrict__ bm, const float* __restrict__ Wsd,
                  const float* __restrict__ bsd, float* __restrict__ Xout,
                  int dil) {
    __shared__ __bf16 sA[2][MST_BM][40];
    __shared__ __bf16 sB[2][32][72];
    const int m0 = blockIdx.x * MST_BM;
    const int bb = blockIdx.y;
    const int wave = threadIdx.x >> 5, lane = threadIdx.x & 31;
    const int tid = threadIdx.x;
    const float* Xb = Xin + (size_t)bb * CC * TT;

    v8f acc[4];
#pragma unroll
    for (int nt = 0; nt < 4; ++nt) acc[nt] = vzero8();

    auto stageA = [&](int buf, int k0) {
        // 128 rows x 64B, 16B per lane-op: 512 chunks over 256 threads
#pragma unroll
        for (int s = 0; s < 2; ++s) {
            int idx = tid + s * 256;
            int r = idx >> 2, p = idx & 3;
            const __bf16* g = wbf + (size_t)(m0 + r) * 4096 + k0 + p * 8;
            async_copy16(&sA[buf][r][p * 8], g);
        }
    };
    auto stageB = [&](int buf, int k0) {
        for (int i = tid; i < 32 * 64; i += 256) {
            int kk = i >> 6, t = i & 63;
            int kg = k0 + kk;
            int ci = kg & (CC - 1);
            float vv;
            if (kg < CC) vv = Xb[(size_t)ci * TT + t];
            else         vv = (t >= dil) ? Xb[(size_t)ci * TT + t - dil] : 0.f;
            sB[buf][kk][t] = tob(vv);
        }
    };

    stageA(0, 0);
    stageB(0, 0);
    wait_async0();
    __syncthreads();

    int cur = 0;
    for (int it = 0; it < MST_ITERS; ++it) {
        if (it + 1 < MST_ITERS) {           // prefetch next tiles into nxt buffer
            stageA(cur ^ 1, (it + 1) * 32);
            stageB(cur ^ 1, (it + 1) * 32);
        }
        v16bf a = load_frag_a(&sA[cur][wave * 16][0], 40);
#pragma unroll
        for (int nt = 0; nt < 4; ++nt) {
            v16bf bfr = load_frag_b(&sB[cur][0][0], 72, nt * 16);
            acc[nt] = wmma_bf16(a, bfr, acc[nt]);
        }
        wait_async0();
        __syncthreads();
        cur ^= 1;
    }

    const int m = lane & 15, h = lane >> 4;
#pragma unroll
    for (int nt = 0; nt < 4; ++nt)
#pragma unroll
        for (int r = 0; r < 8; ++r) {
            int o = m0 + wave * 16 + r + 8 * h;
            int t = nt * 16 + m;
            float dense = fmaxf(acc[nt][r] + bm[o], 0.f);
            float xcur = Xb[(size_t)o * TT + t];
            float xprv = (t >= dil) ? Xb[(size_t)o * TT + t - dil] : 0.f;
            float dw = fmaxf(Wsd[o * 2 + 1] * xcur + Wsd[o * 2 + 0] * xprv + bsd[o], 0.f);
            Xout[((size_t)bb * CC + o) * TT + t] = dense + dw;
        }
}

// ---------------------------------------------------------------------------
// 9) out[b,t,n,d] = xc[b,c,t] + h_final[b,c,t] + x[b,t,n,d]
// ---------------------------------------------------------------------------
__global__ __launch_bounds__(256)
void final_kernel(const float* __restrict__ xc, const float* __restrict__ hfin,
                  const float* __restrict__ x, float* __restrict__ out) {
    size_t i = (size_t)blockIdx.x * 256 + threadIdx.x;   // B*T*N*D = 1048576
    int d = (int)(i & 1);
    size_t r = i >> 1;
    int n = (int)(r & 1023); r >>= 10;
    int t = (int)(r & 63);
    int bb = (int)(r >> 6);
    size_t ci = ((size_t)bb * CC + (n * 2 + d)) * TT + t;
    out[i] = xc[ci] + hfin[ci] + x[i];
}

// ---------------------------------------------------------------------------
// Launch
// ---------------------------------------------------------------------------
extern "C" void kernel_launch(void* const* d_in, const int* in_sizes, int n_in,
                              void* d_out, int out_size, void* d_ws, size_t ws_size,
                              hipStream_t stream) {
    const float* x     = (const float*)d_in[0];
    const float* graph = (const float*)d_in[1];
    const float* W1 = (const float*)d_in[2];  const float* b1 = (const float*)d_in[3];
    const float* W2 = (const float*)d_in[4];  const float* b2 = (const float*)d_in[5];
    const float* Wq = (const float*)d_in[6];  const float* bq = (const float*)d_in[7];
    const float* Wk = (const float*)d_in[8];  const float* bk = (const float*)d_in[9];
    const float* Wv = (const float*)d_in[10]; const float* bv = (const float*)d_in[11];
    const float* Wo = (const float*)d_in[12]; const float* bo = (const float*)d_in[13];
    const float* Wm[4], *bm[4], *Wsd[4], *bsd[4];
    for (int i = 0; i < 4; ++i) {
        Wm[i]  = (const float*)d_in[14 + 4 * i];
        bm[i]  = (const float*)d_in[15 + 4 * i];
        Wsd[i] = (const float*)d_in[16 + 4 * i];
        bsd[i] = (const float*)d_in[17 + 4 * i];
    }
    float* out = (float*)d_out;

    // workspace layout
    float* ws = (float*)d_ws;
    size_t off = 0;
    float* gbuf  = ws + off; off += (size_t)BB * TT * NN * DD;   // 1M
    float* xc    = ws + off; off += (size_t)BB * CC * TT;        // 1M
    float* hA    = ws + off; off += (size_t)BB * CC * TT;        // 1M
    float* hB    = ws + off; off += (size_t)BB * CC * TT;        // 1M
    float* q_    = ws + off; off += (size_t)BB * NN * 256;       // 2M
    float* k_    = ws + off; off += (size_t)BB * NN * 256;       // 2M
    float* v_    = ws + off; off += (size_t)BB * NN * 128;       // 1M
    float* attnb = ws + off; off += (size_t)BB * NN * 128;       // 1M
    float* kvb   = ws + off; off += (size_t)BB * 256 * 128;      // 256K
    float* ksum  = ws + off; off += (size_t)BB * 256;
    float* zbuf  = ws + off; off += (size_t)BB * NN;
    __bf16* wbf  = (__bf16*)(ws + off);       // 2048*4096 bf16 = 16.8 MB (reused per block)

    // 1) GCN
    gcn_kernel<<<64, 256, 0, stream>>>(x, graph, W1, b1, W2, b2, gbuf);
    // 2) QKV + feature map
    qkv_kernel<<<128, 256, 0, stream>>>(x, Wq, bq, Wk, bk, Wv, bv, q_, k_, v_);
    // 3) ksum, 4) z
    ksum_kernel<<<BB, 256, 0, stream>>>(k_, ksum);
    z_kernel<<<32, 256, 0, stream>>>(q_, ksum, zbuf);
    // 5) kv
    kv_kernel<<<dim3(4, BB), 256, 0, stream>>>(k_, v_, kvb);
    // 6) q_ @ kv (+xf)
    attn1_kernel<<<128, 256, 0, stream>>>(q_, kvb, zbuf, x, attnb);
    // 7) @ Wo + residuals -> xc [B,C,T]
    attn2_kernel<<<128, 256, 0, stream>>>(attnb, Wo, bo, gbuf, x, xc);
    // 8) MSTCN: repack weights to bf16, then dense+depthwise block (x4)
    const int dils[4] = {1, 2, 4, 4};
    const float* src[4] = {xc, hA, hB, hA};
    float*       dst[4] = {hA, hB, hA, hB};
    for (int blk = 0; blk < 4; ++blk) {
        wconv_kernel<<<32768, 256, 0, stream>>>(Wm[blk], wbf);
        mstcn_kernel<<<dim3(CC / MST_BM, BB), 256, 0, stream>>>(
            src[blk], wbf, bm[blk], Wsd[blk], bsd[blk], dst[blk], dils[blk]);
    }
    // 9) output
    final_kernel<<<4096, 256, 0, stream>>>(xc, hB, x, out);
}
